// CondConvEncoder_23407571763754
// MI455X (gfx1250) — compile-verified, matrix operand order checked by reading, not run on hardware
//
#include <hip/hip_runtime.h>

typedef __attribute__((ext_vector_type(16))) _Float16 v16h;
typedef __attribute__((ext_vector_type(8)))  _Float16 v8h;
typedef __attribute__((ext_vector_type(8)))  float    v8f;

#define BATCH 16
#define MAXK  896   // largest padded K (res4.w2: 32*27 -> 896)
#define RED_T 256

// ---------------------------------------------------------------------------
// Mix expert weights per sample: dst[b][m][k] = sum_e rw[b,slot,e] * w[e,m,k]
// dst is f16, zero-padded to [Mpad][Kpad] (zeros make k>=Kreal / m>=O inert).
// ---------------------------------------------------------------------------
__global__ void mix_w_kernel(const float* __restrict__ w, const float* __restrict__ rw,
                             _Float16* __restrict__ dst, int slot, int O, int K,
                             int Mpad, int Kpad) {
    int b = blockIdx.y;
    long long idx = (long long)blockIdx.x * blockDim.x + threadIdx.x;
    long long tot = (long long)Mpad * Kpad;
    if (idx >= tot) return;
    int m = (int)(idx / Kpad);
    int k = (int)(idx - (long long)m * Kpad);
    float acc = 0.f;
    if (m < O && k < K) {
        const float* r = rw + b * 45 + slot * 3;
        long long ok = (long long)O * K;
        long long p  = (long long)m * K + k;
        acc = r[0] * w[p] + r[1] * w[ok + p] + r[2] * w[2 * ok + p];
    }
    dst[(long long)b * tot + idx] = (_Float16)acc;
}

__global__ void mix_b_kernel(const float* __restrict__ wb, const float* __restrict__ rw,
                             float* __restrict__ dst, int slot, int O, int Mpad) {
    int b = blockIdx.x;
    int m = threadIdx.x;
    if (m >= Mpad) return;
    float acc = 0.f;
    if (m < O) {
        const float* r = rw + b * 45 + slot * 3;
        acc = r[0] * wb[m] + r[1] * wb[O + m] + r[2] * wb[2 * O + m];
    }
    dst[b * Mpad + m] = acc;
}

__global__ void f32tof16_kernel(const float* __restrict__ src, _Float16* __restrict__ dst,
                                long long n) {
    long long i = (long long)blockIdx.x * blockDim.x + threadIdx.x;
    if (i < n) dst[i] = (_Float16)src[i];
}

// ---------------------------------------------------------------------------
// Conditional conv3d as implicit GEMM with v_wmma_f32_16x16x32_f16.
// 256 threads = 8 waves; each wave owns one 16(M)x16(N) tile.
// K-decomposition is precomputed once per block into LDS (delta + packed taps).
// Interior tiles (ballot-uniform) take a checkless fast path.
// ---------------------------------------------------------------------------
template <int KK>
__global__ __launch_bounds__(256)
void condconv_wmma(const _Float16* __restrict__ xh, const _Float16* __restrict__ wA,
                   const float* __restrict__ bias, float* __restrict__ out,
                   int Cin, int ID, int IH, int IW,
                   int OD, int OH, int OW,
                   int sd, int sh, int sw, int pad,
                   int Cout, int Mpad, int Kreal, int Kpad) {
    __shared__ int s_delta[MAXK];
    __shared__ int s_pack[MAXK];

    const int IHIW = IH * IW;
    const int CST  = ID * IHIW;
    const int KK2 = KK * KK, KK3 = KK2 * KK;

    for (int k = threadIdx.x; k < Kpad; k += 256) {
        int d = 0, pk = 0x00FFFFFF;            // sentinel: kd=kh=kw=255 -> never valid
        if (k < Kreal) {
            int ci  = k / KK3;                  // constant division (template)
            int rem = k - ci * KK3;
            int kd  = rem / KK2;
            int r2  = rem - kd * KK2;
            int kh  = r2 / KK;
            int kw  = r2 - kh * KK;
            d  = ci * CST + kd * IHIW + kh * IW + kw;
            pk = kd | (kh << 8) | (kw << 16);
        }
        s_delta[k] = d;
        s_pack[k]  = pk;
    }
    __syncthreads();

    const int lane  = threadIdx.x & 31;
    const int wave  = threadIdx.x >> 5;
    const int ntile = blockIdx.x * 8 + wave;
    const int mtile = blockIdx.y;
    const int b     = blockIdx.z;
    const int Npos  = OD * OH * OW;
    const int half  = lane >> 4;
    const int nl    = lane & 15;

    const int pos = ntile * 16 + nl;
    const bool posok = pos < Npos;
    int od = 0, oh = 0, ow = 0;
    if (posok) {
        od = pos / (OH * OW);
        int r = pos - od * (OH * OW);
        oh = r / OW;
        ow = r - oh * OW;
    }
    const int odb = od * sd - pad;
    const int ohb = oh * sh - pad;
    const int owb = ow * sw - pad;
    const int base = odb * IHIW + ohb * IW + owb;

    bool interior = posok &&
                    (odb >= 0) && (odb + KK <= ID) &&
                    (ohb >= 0) && (ohb + KK <= IH) &&
                    (owb >= 0) && (owb + KK <= IW);
    bool allint = (__builtin_amdgcn_ballot_w32(interior) == 0xffffffffu);

    const _Float16* wrow = wA + ((long long)b * Mpad + mtile * 16 + nl) * Kpad;
    const _Float16* xb   = xh + (long long)b * Cin * CST;

    v8f c = {};
    if (allint) {
        for (int k0 = 0; k0 < Kpad; k0 += 32) {
            v16h a, bf;
            v8h lo = *(const v8h*)(wrow + k0 + half * 8);
            v8h hi = *(const v8h*)(wrow + k0 + 16 + half * 8);
#pragma unroll
            for (int i = 0; i < 8; ++i) { a[i] = lo[i]; a[8 + i] = hi[i]; }
#pragma unroll
            for (int v = 0; v < 8; ++v) {
                int kb = (v >> 2) * 16 + half * 8 + (v & 3) * 2;
#pragma unroll
                for (int p = 0; p < 2; ++p) {
                    int kg = k0 + kb + p;
                    bf[2 * v + p] = xb[base + s_delta[kg]];
                }
            }
            c = __builtin_amdgcn_wmma_f32_16x16x32_f16(false, a, false, bf,
                                                       (short)0, c, false, false);
        }
    } else {
        for (int k0 = 0; k0 < Kpad; k0 += 32) {
            v16h a, bf;
            v8h lo = *(const v8h*)(wrow + k0 + half * 8);
            v8h hi = *(const v8h*)(wrow + k0 + 16 + half * 8);
#pragma unroll
            for (int i = 0; i < 8; ++i) { a[i] = lo[i]; a[8 + i] = hi[i]; }
#pragma unroll
            for (int v = 0; v < 8; ++v) {
                int kb = (v >> 2) * 16 + half * 8 + (v & 3) * 2;
#pragma unroll
                for (int p = 0; p < 2; ++p) {
                    int kg = k0 + kb + p;
                    int pk = s_pack[kg];
                    int id = odb + (pk & 0xFF);
                    int ih = ohb + ((pk >> 8) & 0xFF);
                    int iw = owb + ((pk >> 16) & 0xFF);
                    bool valid = posok &
                                 ((unsigned)id < (unsigned)ID) &
                                 ((unsigned)ih < (unsigned)IH) &
                                 ((unsigned)iw < (unsigned)IW);
                    int addr = valid ? (base + s_delta[kg]) : 0;
                    _Float16 val = xb[addr];
                    bf[2 * v + p] = valid ? val : (_Float16)0.f;
                }
            }
            c = __builtin_amdgcn_wmma_f32_16x16x32_f16(false, a, false, bf,
                                                       (short)0, c, false, false);
        }
    }

#pragma unroll
    for (int r = 0; r < 8; ++r) {
        int m = mtile * 16 + r + 8 * half;
        int n = ntile * 16 + nl;
        if (m < Cout && n < Npos) {
            out[((long long)b * Cout + m) * Npos + n] = c[r] + bias[b * Mpad + m];
        }
    }
}

// ---------------------------------------------------------------------------
// Deterministic per-channel batch statistics (no divisions, no atomics).
// ---------------------------------------------------------------------------
__global__ void bn_partial(const float* __restrict__ x, float* __restrict__ psum,
                           float* __restrict__ psq, int C, int Npos, int Bn, int nchunk) {
    int c = blockIdx.x;
    int chunk = blockIdx.y;
    float s = 0.f, sq = 0.f;
    for (int b = 0; b < Bn; ++b) {
        const float* p = x + ((long long)b * C + c) * Npos;
        for (int pos = chunk * RED_T + threadIdx.x; pos < Npos; pos += nchunk * RED_T) {
            float v = p[pos];
            s += v;
            sq += v * v;
        }
    }
    __shared__ float ss[RED_T], ssq[RED_T];
    ss[threadIdx.x] = s;
    ssq[threadIdx.x] = sq;
    __syncthreads();
    for (int st = RED_T / 2; st > 0; st >>= 1) {
        if ((int)threadIdx.x < st) {
            ss[threadIdx.x] += ss[threadIdx.x + st];
            ssq[threadIdx.x] += ssq[threadIdx.x + st];
        }
        __syncthreads();
    }
    if (threadIdx.x == 0) {
        psum[c * nchunk + chunk] = ss[0];
        psq[c * nchunk + chunk]  = ssq[0];
    }
}

__global__ void bn_finalize(const float* __restrict__ psum, const float* __restrict__ psq,
                            const float* __restrict__ g, const float* __restrict__ be,
                            float* __restrict__ scsh, int nchunk, long long cnt) {
    int c = blockIdx.x;
    if (threadIdx.x != 0) return;
    float s = 0.f, sq = 0.f;
    for (int i = 0; i < nchunk; ++i) {
        s += psum[c * nchunk + i];
        sq += psq[c * nchunk + i];
    }
    float m   = s / (float)cnt;
    float var = sq / (float)cnt - m * m;
    float sc  = g[c] * rsqrtf(var + 1e-5f);
    scsh[2 * c]     = sc;
    scsh[2 * c + 1] = be[c] - m * sc;
}

// normalize + leakyReLU(0.2) (+ optional residual add); writes f32 in place
// and an f16 mirror for the downstream WMMA consumer. Channel from blockIdx.
__global__ void bn_act(float* __restrict__ x, _Float16* __restrict__ xm,
                       const float* __restrict__ scsh, const float* __restrict__ addsrc,
                       int C, int Npos) {
    int bc = blockIdx.y;
    int c  = bc - (bc / C) * C;
    int pos = blockIdx.x * 256 + threadIdx.x;
    if (pos >= Npos) return;
    long long i = (long long)bc * Npos + pos;
    float y = scsh[2 * c] * x[i] + scsh[2 * c + 1];
    y = (y >= 0.f) ? y : 0.2f * y;
    if (addsrc) y += addsrc[i];
    x[i]  = y;
    xm[i] = (_Float16)y;
}

// ---------------------------------------------------------------------------
// Host orchestration
// ---------------------------------------------------------------------------
extern "C" void kernel_launch(void* const* d_in, const int* in_sizes, int n_in,
                              void* d_out, int out_size, void* d_ws, size_t ws_size,
                              hipStream_t stream) {
    (void)in_sizes; (void)n_in; (void)out_size; (void)ws_size;
    const float* X  = (const float*)d_in[0];
    const float* RW = (const float*)d_in[1];
    auto in = [&](int i) { return (const float*)d_in[i]; };

    struct CV { int slot, wi, bi, O, I, kk; };
    const CV cv[15] = {
        { 0,  2,  3,   8,  4, 3},   // conv1
        { 1,  6,  7,   8,  8, 3},   // res1.w1
        { 2,  8,  9,   8,  8, 3},   // res1.w2
        { 3, 10, 11,   8,  8, 3},   // res1.w3
        { 4, 18, 19,   8,  8, 3},   // res2.w1
        { 5, 20, 21,  16,  8, 3},   // res2.w2
        { 6, 22, 23,  16,  8, 3},   // res2.w3
        { 7, 30, 31,  16, 16, 3},   // res3.w1
        { 8, 32, 33,  32, 16, 3},   // res3.w2
        { 9, 34, 35,  32, 16, 3},   // res3.w3
        {10, 42, 43,  32, 32, 3},   // res4.w1
        {11, 44, 45,  64, 32, 3},   // res4.w2
        {12, 46, 47,  64, 32, 3},   // res4.w3
        {13, 54, 55, 256, 64, 1},   // latent
        {14, 56, 57, 256, 64, 1},   // latent_std
    };

    char* base = (char*)d_ws;
    size_t cur = 0;
    auto alloc = [&](size_t bytes) -> void* {
        size_t off = (cur + 255) & ~(size_t)255;
        cur = off + bytes;
        return (void*)(base + off);
    };

    _Float16* wb[15]; float* bb[15]; int Mp[15], Kp[15], Kr[15];
    for (int i = 0; i < 15; ++i) {
        int K = cv[i].I * cv[i].kk * cv[i].kk * cv[i].kk;
        Kr[i] = K;
        Kp[i] = ((K + 31) / 32) * 32;
        Mp[i] = ((cv[i].O + 15) / 16) * 16;
        wb[i] = (_Float16*)alloc((size_t)BATCH * Mp[i] * Kp[i] * sizeof(_Float16));
        bb[i] = (float*)alloc((size_t)BATCH * Mp[i] * sizeof(float));
    }
    // f32 buffers + f16 mirrors
    auto fbuf = [&](size_t elems) { return (float*)alloc(elems * 4); };
    auto hbuf = [&](size_t elems) { return (_Float16*)alloc(elems * 2); };
    size_t nh0 = (size_t)16 * 8 * 262144;
    size_t nr1 = (size_t)16 * 8 * 32768;
    size_t nr2a = (size_t)16 * 8 * 4096, nr2 = (size_t)16 * 16 * 4096;
    size_t nr3a = (size_t)16 * 16 * 512, nr3 = (size_t)16 * 32 * 512;
    size_t nr4a = (size_t)16 * 32 * 128, nr4 = (size_t)16 * 64 * 128;
    size_t nx  = (size_t)16 * 4 * 262144;

    float *h0 = fbuf(nh0), *r1a = fbuf(nr1), *r1b = fbuf(nr1), *r1s = fbuf(nr1);
    float *r2a = fbuf(nr2a), *r2b = fbuf(nr2), *r2s = fbuf(nr2);
    float *r3a = fbuf(nr3a), *r3b = fbuf(nr3), *r3s = fbuf(nr3);
    float *r4a = fbuf(nr4a), *r4b = fbuf(nr4), *r4s = fbuf(nr4);
    _Float16 *xhm = hbuf(nx);
    _Float16 *h0m = hbuf(nh0), *r1am = hbuf(nr1), *r1bm = hbuf(nr1), *r1sm = hbuf(nr1);
    _Float16 *r2am = hbuf(nr2a), *r2bm = hbuf(nr2), *r2sm = hbuf(nr2);
    _Float16 *r3am = hbuf(nr3a), *r3bm = hbuf(nr3), *r3sm = hbuf(nr3);
    _Float16 *r4am = hbuf(nr4a), *r4bm = hbuf(nr4), *r4sm = hbuf(nr4);
    float* psum = fbuf((size_t)64 * 256);
    float* psq  = fbuf((size_t)64 * 256);
    float* scsh = fbuf((size_t)128);

    // 1) mix all expert weights + biases; convert x to f16
    for (int i = 0; i < 15; ++i) {
        long long tot = (long long)Mp[i] * Kp[i];
        dim3 g((unsigned)((tot + 255) / 256), BATCH);
        mix_w_kernel<<<g, 256, 0, stream>>>(in(cv[i].wi), RW, wb[i], cv[i].slot,
                                            cv[i].O, Kr[i], Mp[i], Kp[i]);
        mix_b_kernel<<<dim3(BATCH), Mp[i], 0, stream>>>(in(cv[i].bi), RW, bb[i],
                                                        cv[i].slot, cv[i].O, Mp[i]);
    }
    f32tof16_kernel<<<dim3((unsigned)((nx + 255) / 256)), 256, 0, stream>>>(X, xhm,
                                                                            (long long)nx);

    auto gemm = [&](const _Float16* srcH, int i, float* dst, int Cin,
                    int ID, int IH, int IW, int OD, int OH, int OW,
                    int sd, int sh, int sw, int pad) {
        int Npos = OD * OH * OW;
        dim3 grid((unsigned)((Npos + 127) / 128), (unsigned)(Mp[i] / 16), BATCH);
        if (cv[i].kk == 3) {
            condconv_wmma<3><<<grid, 256, 0, stream>>>(srcH, wb[i], bb[i], dst,
                Cin, ID, IH, IW, OD, OH, OW, sd, sh, sw, pad,
                cv[i].O, Mp[i], Kr[i], Kp[i]);
        } else {
            condconv_wmma<1><<<grid, 256, 0, stream>>>(srcH, wb[i], bb[i], dst,
                Cin, ID, IH, IW, OD, OH, OW, sd, sh, sw, pad,
                cv[i].O, Mp[i], Kr[i], Kp[i]);
        }
    };

    auto bnact = [&](float* buf, _Float16* mir, const float* g, const float* be,
                     int C, int Npos, const float* add) {
        long long total = (long long)BATCH * C * Npos;
        long long chunks = total / (256 * 64);
        if (chunks < 1) chunks = 1;
        if (chunks > 256) chunks = 256;
        int nchunk = (int)chunks;
        bn_partial<<<dim3((unsigned)C, (unsigned)nchunk), RED_T, 0, stream>>>(
            buf, psum, psq, C, Npos, BATCH, nchunk);
        bn_finalize<<<dim3((unsigned)C), 32, 0, stream>>>(psum, psq, g, be, scsh,
                                                          nchunk, (long long)BATCH * Npos);
        bn_act<<<dim3((unsigned)((Npos + 255) / 256), (unsigned)(BATCH * C)), 256, 0,
                 stream>>>(buf, mir, scsh, add, C, Npos);
    };

    // conv1: [16,4,64^3] -> [16,8,64^3]
    gemm(xhm, 0, h0, 4, 64, 64, 64, 64, 64, 64, 1, 1, 1, 1);
    bnact(h0, h0m, in(4), in(5), 8, 262144, nullptr);

    // res1: 8 -> 8, stride 2  (64^3 -> 32^3)
    gemm(h0m, 1, r1a, 8, 64, 64, 64, 32, 32, 32, 2, 2, 2, 1);
    bnact(r1a, r1am, in(12), in(13), 8, 32768, nullptr);
    gemm(r1am, 2, r1b, 8, 32, 32, 32, 32, 32, 32, 1, 1, 1, 1);
    gemm(h0m, 3, r1s, 8, 64, 64, 64, 32, 32, 32, 2, 2, 2, 1);
    bnact(r1s, r1sm, in(16), in(17), 8, 32768, nullptr);
    bnact(r1b, r1bm, in(14), in(15), 8, 32768, r1s);

    // res2: 8 -> 16, stride 2  (32^3 -> 16^3)
    gemm(r1bm, 4, r2a, 8, 32, 32, 32, 16, 16, 16, 2, 2, 2, 1);
    bnact(r2a, r2am, in(24), in(25), 8, 4096, nullptr);
    gemm(r2am, 5, r2b, 8, 16, 16, 16, 16, 16, 16, 1, 1, 1, 1);
    gemm(r1bm, 6, r2s, 8, 32, 32, 32, 16, 16, 16, 2, 2, 2, 1);
    bnact(r2s, r2sm, in(28), in(29), 16, 4096, nullptr);
    bnact(r2b, r2bm, in(26), in(27), 16, 4096, r2s);

    // res3: 16 -> 32, stride 2  (16^3 -> 8^3)
    gemm(r2bm, 7, r3a, 16, 16, 16, 16, 8, 8, 8, 2, 2, 2, 1);
    bnact(r3a, r3am, in(36), in(37), 16, 512, nullptr);
    gemm(r3am, 8, r3b, 16, 8, 8, 8, 8, 8, 8, 1, 1, 1, 1);
    gemm(r2bm, 9, r3s, 16, 16, 16, 16, 8, 8, 8, 2, 2, 2, 1);
    bnact(r3s, r3sm, in(40), in(41), 32, 512, nullptr);
    bnact(r3b, r3bm, in(38), in(39), 32, 512, r3s);

    // res4: 32 -> 64, stride (2,2,1)  ([8,8,8] -> [4,4,8])
    gemm(r3bm, 10, r4a, 32, 8, 8, 8, 4, 4, 8, 2, 2, 1, 1);
    bnact(r4a, r4am, in(48), in(49), 32, 128, nullptr);
    gemm(r4am, 11, r4b, 32, 4, 4, 8, 4, 4, 8, 1, 1, 1, 1);
    gemm(r3bm, 12, r4s, 32, 8, 8, 8, 4, 4, 8, 2, 2, 1, 1);
    bnact(r4s, r4sm, in(52), in(53), 64, 128, nullptr);
    bnact(r4b, r4bm, in(50), in(51), 64, 128, r4s);

    // latent heads: 1x1x1, 64 -> 256, straight into d_out
    float* OUT = (float*)d_out;
    gemm(r4bm, 13, OUT,          64, 4, 4, 8, 4, 4, 8, 1, 1, 1, 0);
    gemm(r4bm, 14, OUT + 524288, 64, 4, 4, 8, 4, 4, 8, 1, 1, 1, 0);
}
